// PointNet2SAMSGSampling_32126355374238
// MI455X (gfx1250) — compile-verified, matrix operand order checked by reading, not run on hardware
//
#include <hip/hip_runtime.h>

#define NPTS 16384
#define NB   4
#define NQ   1024
#define BN_SCALE 0.99999500003749968750f  // 1/sqrt(1+1e-5)

typedef __attribute__((ext_vector_type(16))) _Float16 v16h;
typedef __attribute__((ext_vector_type(8)))  _Float16 v8h;   // 16B, naturally aligned
typedef __attribute__((ext_vector_type(8)))  float    v8f;

// ---------- WMMA helpers (f16 16x16x32, f32 accum) ----------
__device__ __forceinline__ int kmap_i(int i, int half) {
  // v16h element i -> K offset within a 32-wide K slab (ISA 7.12.2 16-bit layout)
  return (i < 8) ? (8 * half + i) : (16 + 8 * half + (i - 8));
}

__device__ __forceinline__ v8f wmma16(v16h a, v16h b, v8f c) {
  return __builtin_amdgcn_wmma_f32_16x16x32_f16(false, a, false, b, (short)0, c,
                                                false, false);
}

__device__ __forceinline__ v8f vzero() {
  v8f z;
#pragma unroll
  for (int e = 0; e < 8; ++e) z[e] = 0.f;
  return z;
}

__device__ __forceinline__ v16h combine8(v8h lo, v8h hi) {
  return __builtin_shufflevector(lo, hi, 0, 1, 2, 3, 4, 5, 6, 7, 8, 9, 10, 11,
                                 12, 13, 14, 15);
}

// A fragment: two contiguous 16B runs per lane -> 2x b128 loads
__device__ __forceinline__ v16h load_a_frag(const _Float16* X, int stride,
                                            int m0, int k0, int lane) {
  int half = lane >> 4;
  const _Float16* p = X + (m0 + (lane & 15)) * stride + k0 + 8 * half;
  return combine8(*(const v8h*)p, *(const v8h*)(p + 16));
}

// B fragment from pre-swizzled weight buffer: 32B contiguous per lane
__device__ __forceinline__ v16h load_b_frag(const _Float16* Wf, int frag,
                                            int lane) {
  const _Float16* p = Wf + ((frag * 32 + lane) << 4);
  return combine8(*(const v8h*)p, *(const v8h*)(p + 8));
}

// Stage weights (cout x cin, row-major f32) into fragment-order f16 LDS:
// sW[(kt*coutTiles+nt)*512 + lane*16 + i] = WT[k][col] with zero-pad k>=cin
__device__ __forceinline__ void stage_wfrag(_Float16* sW,
                                            const float* __restrict__ W,
                                            int cin, int ktTiles, int coutTiles,
                                            int tid, int nthreads) {
  int total = ktTiles * coutTiles * 512;
  for (int t = tid; t < total; t += nthreads) {
    int frag = t >> 9;
    int within = t & 511;
    int lane = within >> 4;
    int i = within & 15;
    int kt = frag / coutTiles, nt = frag - kt * coutTiles;
    int half = lane >> 4;
    int k = kt * 32 + kmap_i(i, half);
    int col = nt * 16 + (lane & 15);
    sW[t] = (_Float16)((k < cin) ? W[col * cin + k] : 0.f);
  }
}

template <int KT, int NT>
__device__ __forceinline__ void tile_mm(const _Float16* A, int astride, int m0,
                                        const _Float16* Wf, int lane,
                                        v8f* acc) {
#pragma unroll
  for (int kt = 0; kt < KT; ++kt) {
    v16h a = load_a_frag(A, astride, m0, kt * 32, lane);
#pragma unroll
    for (int nt = 0; nt < NT; ++nt) {
      v16h bb = load_b_frag(Wf, kt * NT + nt, lane);
      acc[nt] = wmma16(a, bb, acc[nt]);
    }
  }
}

template <int NT>
__device__ __forceinline__ void store_relu(_Float16* Y, int stride, int m0,
                                           int lane, const v8f* acc,
                                           const float* sS, const float* sB) {
  int half = lane >> 4, col = lane & 15;
#pragma unroll
  for (int nt = 0; nt < NT; ++nt) {
    int c = nt * 16 + col;
    float s = sS[c], o = sB[c];
#pragma unroll
    for (int j = 0; j < 8; ++j) {
      float v = acc[nt][j] * s + o;
      Y[(m0 + j + 8 * half) * stride + c] = (_Float16)(v > 0.f ? v : 0.f);
    }
  }
}

// ---------- Kernel 1: confidence MLP (64->64 WMMA, 64->3 VALU) + keys ----------
__global__ __launch_bounds__(128) void conf_kernel(
    const float* __restrict__ feats, const float* __restrict__ w1,
    const float* __restrict__ g1, const float* __restrict__ b1,
    const float* __restrict__ w2, float* __restrict__ cls_out,
    unsigned* __restrict__ keys) {
  __shared__ _Float16 sW[2 * 4 * 512];  // fragment-order 64x64
  __shared__ float sS[64], sB[64], sW2[3 * 64];
  __shared__ float sH[4][16][64];
  int tid = threadIdx.x, lane = tid & 31, w = tid >> 5;
  stage_wfrag(sW, w1, 64, 2, 4, tid, 128);
  if (tid < 64) { sS[tid] = g1[tid] * BN_SCALE; sB[tid] = b1[tid]; }
  for (int t = tid; t < 192; t += 128) sW2[t] = w2[t];
  __syncthreads();

  int tile = blockIdx.x * 4 + w;
  int r0 = tile * 16;
  int b = r0 >> 14, n0 = r0 & (NPTS - 1);
  const float* F = feats + (size_t)b * 64 * NPTS;
  int half = lane >> 4, col = lane & 15;

  v8f acc[4];
#pragma unroll
  for (int nt = 0; nt < 4; ++nt) acc[nt] = vzero();
#pragma unroll
  for (int kt = 0; kt < 2; ++kt) {
    v16h a;  // ft[n][c] = features[b][c][n] (strided gather, small kernel)
#pragma unroll
    for (int i = 0; i < 16; ++i) {
      int kb = kt * 32 + kmap_i(i, half);
      a[i] = (_Float16)F[(size_t)kb * NPTS + n0 + col];
    }
#pragma unroll
    for (int nt = 0; nt < 4; ++nt) {
      v16h bb = load_b_frag(sW, kt * 4 + nt, lane);
      acc[nt] = wmma16(a, bb, acc[nt]);
    }
  }
#pragma unroll
  for (int nt = 0; nt < 4; ++nt) {
    int c = nt * 16 + col;
    float s = sS[c], o = sB[c];
#pragma unroll
    for (int j = 0; j < 8; ++j) {
      float v = acc[nt][j] * s + o;
      sH[w][j + 8 * half][c] = v > 0.f ? v : 0.f;
    }
  }
  __syncthreads();
  if (lane < 16) {
    int n = n0 + lane;
    float m = -3.4e38f;
#pragma unroll
    for (int j = 0; j < 3; ++j) {
      float d = 0.f;
      for (int c = 0; c < 64; ++c) d += sH[w][lane][c] * sW2[j * 64 + c];
      cls_out[((size_t)b * 3 + j) * NPTS + n] = d;
      m = fmaxf(m, d);
    }
    unsigned u = __float_as_uint(m);  // sigmoid monotone -> sort by m directly
    keys[b * NPTS + n] = (u & 0x80000000u) ? ~u : (u | 0x80000000u);
  }
}

// ---------- Kernel 2: per-batch radix top-1024 select + bitonic order ----------
__global__ __launch_bounds__(1024) void select_kernel(
    const float* __restrict__ points, const unsigned* __restrict__ keys,
    int* __restrict__ sidx, float* __restrict__ sampled) {
  int b = blockIdx.x, tid = threadIdx.x;
  const int N = NPTS, K = NQ;
  __shared__ int hist[256];
  __shared__ unsigned sprefix;
  __shared__ int skneed;
  __shared__ int countG;
  __shared__ unsigned long long comp[NQ];
  __shared__ int tieCnt[NQ];

  unsigned prefix = 0;
  int kneed = K;
  for (int d = 0; d < 4; ++d) {
    int shift = 24 - 8 * d;
    if (tid < 256) hist[tid] = 0;
    __syncthreads();
    for (int i = tid; i < N; i += 1024) {
      unsigned key = keys[b * N + i];
      bool ok = (d == 0) || ((key >> (shift + 8)) == prefix);
      if (ok) atomicAdd(&hist[(key >> shift) & 255], 1);
    }
    __syncthreads();
    if (tid == 0) {
      int c = 0; unsigned v = 0;
      for (int bin = 255; bin >= 0; --bin) {
        if (c + hist[bin] >= kneed) { v = (unsigned)bin; break; }
        c += hist[bin];
      }
      sprefix = (prefix << 8) | v;
      skneed = kneed - c;
    }
    __syncthreads();
    prefix = sprefix; kneed = skneed;
    __syncthreads();
  }
  unsigned T = prefix;

  if (tid == 0) countG = 0;
  __syncthreads();
  const int chunk = N / 1024;
  int n0 = tid * chunk, localTies = 0;
  for (int i = 0; i < chunk; ++i) {
    unsigned key = keys[b * N + n0 + i];
    if (key > T) {
      int slot = atomicAdd(&countG, 1);
      if (slot < K)
        comp[slot] =
            ((unsigned long long)key << 32) | (unsigned)(0xFFFFFFFFu - (n0 + i));
    } else if (key == T) {
      localTies++;
    }
  }
  tieCnt[tid] = localTies;
  __syncthreads();
  for (int off = 1; off < 1024; off <<= 1) {  // inclusive scan
    int v = (tid >= off) ? tieCnt[tid - off] : 0;
    __syncthreads();
    tieCnt[tid] += v;
    __syncthreads();
  }
  int tieBase = tieCnt[tid] - localTies;
  int cg = countG;
  int takeTies = K - cg;  // ties taken smallest-index first (matches top_k)
  int r = tieBase;
  for (int i = 0; i < chunk; ++i) {
    unsigned key = keys[b * N + n0 + i];
    if (key == T) {
      if (r < takeTies && cg + r < K)
        comp[cg + r] =
            ((unsigned long long)key << 32) | (unsigned)(0xFFFFFFFFu - (n0 + i));
      r++;
    }
  }
  __syncthreads();
  // bitonic sort descending (key desc; tie -> smaller idx first via ~idx)
  for (int ks = 2; ks <= NQ; ks <<= 1) {
    for (int j = ks >> 1; j > 0; j >>= 1) {
      int ixj = tid ^ j;
      if (ixj > tid) {
        bool desc = ((tid & ks) == 0);
        unsigned long long a = comp[tid], c = comp[ixj];
        bool sw = desc ? (a < c) : (a > c);
        if (sw) { comp[tid] = c; comp[ixj] = a; }
      }
      __syncthreads();
    }
  }
  unsigned long long e = comp[tid];
  int idx = (int)(0xFFFFFFFFu - (unsigned)(e & 0xFFFFFFFFu));
  sidx[b * K + tid] = idx;
#pragma unroll
  for (int d = 0; d < 3; ++d)
    sampled[((size_t)b * K + tid) * 3 + d] = points[((size_t)b * N + idx) * 3 + d];
}

// ---------- Kernel 3: ball query, one wave per query, ordered compaction ----------
__global__ __launch_bounds__(256) void ball_kernel(
    const float* __restrict__ points, const float* __restrict__ sampled,
    int* __restrict__ gidx, int nsamp, float r2) {
  int lane = threadIdx.x & 31, w = threadIdx.x >> 5;
  int q = blockIdx.x * 8 + w;
  int b = q >> 10;
  const float* P = points + (size_t)b * NPTS * 3;
  float qx = sampled[q * 3], qy = sampled[q * 3 + 1], qz = sampled[q * 3 + 2];
  int cnt = 0, firstIdx = -1;
  for (int base = 0; base < NPTS && cnt < nsamp; base += 32) {
    int i = base + lane;
    float dx = P[i * 3] - qx, dy = P[i * 3 + 1] - qy, dz = P[i * 3 + 2] - qz;
    bool hit = (dx * dx + dy * dy + dz * dz) < r2;
    unsigned mask = (unsigned)__ballot(hit);
    int rank = __popc(mask & ((1u << lane) - 1u));
    int slot = cnt + rank;
    if (hit && slot < nsamp) gidx[(size_t)q * nsamp + slot] = i;
    if (firstIdx < 0 && mask) {
      int fl = __ffs(mask) - 1;
      firstIdx = __shfl(i, fl, 32);
    }
    cnt += __popc(mask);
  }
  if (cnt < nsamp) {
    if (firstIdx < 0) firstIdx = 0;  // cannot happen: query point is in its ball
    for (int s = cnt + lane; s < nsamp; s += 32) gidx[(size_t)q * nsamp + s] = firstIdx;
  }
}

// ---------- Kernel 4: per-branch gather + 3x(1x1conv+BN+ReLU) + maxpool ----------
template <int NS, int C2>
__global__ __launch_bounds__(128) void branch_kernel(
    const float* __restrict__ points, const float* __restrict__ feats,
    const float* __restrict__ sampled, const int* __restrict__ gidx,
    const float* __restrict__ W1, const float* __restrict__ W2,
    const float* __restrict__ W3, const float* __restrict__ G1,
    const float* __restrict__ Bb1, const float* __restrict__ G2,
    const float* __restrict__ Bb2, const float* __restrict__ G3,
    const float* __restrict__ Bb3, _Float16* __restrict__ cat, int catOff) {
  extern __shared__ char smem[];
  _Float16* sW = (_Float16*)smem;              // <= 12288 halves, fragment order
  float* sS = (float*)(smem + 24576);          // 128
  float* sB = (float*)(smem + 25088);          // 128
  _Float16* sAct = (_Float16*)(smem + 25600);  // per-wave 2 * NS*96 halves

  int tid = threadIdx.x, lane = tid & 31, w = tid >> 5;
  _Float16* bufA = sAct + w * (2 * NS * 96);
  _Float16* bufB = bufA + NS * 96;
  int q = blockIdx.x * 4 + w;
  int b = q >> 10;

  // stage layer1 weights: cin=67 (pad to 96), KT=3, NT=4
  stage_wfrag(sW, W1, 67, 3, 4, tid, 128);
  if (tid < 64) { sS[tid] = G1[tid] * BN_SCALE; sB[tid] = Bb1[tid]; }

  // gather [rel xyz | 64 feats | pad] into bufA (NS x 96, f16)
  float qx = sampled[q * 3], qy = sampled[q * 3 + 1], qz = sampled[q * 3 + 2];
  for (int s = 0; s < NS; ++s) {
    int idx = gidx[(size_t)q * NS + s];
    if (lane < 3) {
      float qc = lane == 0 ? qx : (lane == 1 ? qy : qz);
      bufA[s * 96 + lane] =
          (_Float16)(points[((size_t)b * NPTS + idx) * 3 + lane] - qc);
    }
#pragma unroll
    for (int t = 0; t < 2; ++t) {
      int c = lane + 32 * t;
      bufA[s * 96 + 3 + c] = (_Float16)feats[((size_t)b * 64 + c) * NPTS + idx];
    }
    if (lane < 29) bufA[s * 96 + 67 + lane] = (_Float16)0.f;
  }
  __syncthreads();

  // layer1: (NS x 96) x (96 -> 64)
#pragma unroll
  for (int mt = 0; mt < NS / 16; ++mt) {
    v8f acc[4];
#pragma unroll
    for (int nt = 0; nt < 4; ++nt) acc[nt] = vzero();
    tile_mm<3, 4>(bufA, 96, mt * 16, sW, lane, acc);
    store_relu<4>(bufB, 96, mt * 16, lane, acc, sS, sB);
  }
  __syncthreads();

  // stage layer2 weights: cin=64, KT=2, NT=C2/16
  stage_wfrag(sW, W2, 64, 2, C2 / 16, tid, 128);
  if (tid < C2) { sS[tid] = G2[tid] * BN_SCALE; sB[tid] = Bb2[tid]; }
  __syncthreads();

  // layer2: (NS x 64) x (64 -> C2)
#pragma unroll
  for (int mt = 0; mt < NS / 16; ++mt) {
    v8f acc[C2 / 16];
#pragma unroll
    for (int nt = 0; nt < C2 / 16; ++nt) acc[nt] = vzero();
    tile_mm<2, C2 / 16>(bufB, 96, mt * 16, sW, lane, acc);
    store_relu<C2 / 16>(bufA, 96, mt * 16, lane, acc, sS, sB);
  }
  __syncthreads();

  // stage layer3 weights: cin=C2, KT=C2/32, NT=8
  stage_wfrag(sW, W3, C2, C2 / 32, 8, tid, 128);
  if (tid < 128) { sS[tid] = G3[tid] * BN_SCALE; sB[tid] = Bb3[tid]; }
  __syncthreads();

  // layer3: (NS x C2) x (C2 -> 128), fused maxpool over NS
  int half = lane >> 4, col = lane & 15;
  float cm[8];
#pragma unroll
  for (int nt = 0; nt < 8; ++nt) cm[nt] = 0.f;  // relu outputs are >= 0
#pragma unroll
  for (int mt = 0; mt < NS / 16; ++mt) {
    v8f acc[8];
#pragma unroll
    for (int nt = 0; nt < 8; ++nt) acc[nt] = vzero();
    tile_mm<C2 / 32, 8>(bufA, 96, mt * 16, sW, lane, acc);
#pragma unroll
    for (int nt = 0; nt < 8; ++nt) {
      int c = nt * 16 + col;
      float s = sS[c], o = sB[c];
#pragma unroll
      for (int j = 0; j < 8; ++j) {
        float v = acc[nt][j] * s + o;
        v = v > 0.f ? v : 0.f;
        cm[nt] = fmaxf(cm[nt], v);
      }
    }
  }
#pragma unroll
  for (int nt = 0; nt < 8; ++nt) {
    float o = __shfl_xor(cm[nt], 16, 32);  // merge half-wave row groups
    cm[nt] = fmaxf(cm[nt], o);
  }
  if (lane < 16) {
#pragma unroll
    for (int nt = 0; nt < 8; ++nt)
      cat[(size_t)q * 384 + catOff + nt * 16 + lane] = (_Float16)cm[nt];
  }
}

// ---------- Kernel 5: aggregation GEMM (4096 x 384 -> 128) + BN + ReLU ----------
__global__ __launch_bounds__(128) void agg_kernel(
    const _Float16* __restrict__ cat, const float* __restrict__ Wg,
    const float* __restrict__ Gg, const float* __restrict__ Bg,
    float* __restrict__ y) {
  extern __shared__ char smem[];
  _Float16* sW = (_Float16*)smem;  // fragment order [12*8*512] = 96KB
  float* sS = (float*)(smem + 98304);
  float* sB = (float*)(smem + 98816);
  int tid = threadIdx.x, lane = tid & 31, w = tid >> 5;
  stage_wfrag(sW, Wg, 384, 12, 8, tid, 128);
  if (tid < 128) { sS[tid] = Gg[tid] * BN_SCALE; sB[tid] = Bg[tid]; }
  __syncthreads();

  int tile = blockIdx.x * 4 + w;
  int r0 = tile * 16;
  v8f acc[8];
#pragma unroll
  for (int nt = 0; nt < 8; ++nt) acc[nt] = vzero();
#pragma unroll
  for (int kt = 0; kt < 12; ++kt) {
    v16h a = load_a_frag(cat + (size_t)r0 * 384, 384, 0, kt * 32, lane);
#pragma unroll
    for (int nt = 0; nt < 8; ++nt) {
      v16h bb = load_b_frag(sW, kt * 8 + nt, lane);
      acc[nt] = wmma16(a, bb, acc[nt]);
    }
  }
  int half = lane >> 4, col = lane & 15;
#pragma unroll
  for (int nt = 0; nt < 8; ++nt) {
    int c = nt * 16 + col;
    float s = sS[c], o = sB[c];
#pragma unroll
    for (int j = 0; j < 8; ++j) {
      float v = acc[nt][j] * s + o;
      v = v > 0.f ? v : 0.f;
      int rr = r0 + j + 8 * half;
      int b = rr >> 10, kq = rr & 1023;
      y[((size_t)b * 128 + c) * 1024 + kq] = v;  // (B,128,K) layout
    }
  }
}

// ---------- host launch ----------
extern "C" void kernel_launch(void* const* d_in, const int* in_sizes, int n_in,
                              void* d_out, int out_size, void* d_ws,
                              size_t ws_size, hipStream_t stream) {
  (void)in_sizes; (void)n_in; (void)out_size; (void)ws_size;
  const float* points  = (const float*)d_in[0];
  const float* feats   = (const float*)d_in[1];
  const float* w_conf1 = (const float*)d_in[2];
  const float* g_conf1 = (const float*)d_in[3];
  const float* b_conf1 = (const float*)d_in[4];
  const float* w_conf2 = (const float*)d_in[5];
  const float *bw[9], *bg[9], *bb[9];
  for (int i = 0; i < 9; ++i) {
    bw[i] = (const float*)d_in[6 + i];
    bg[i] = (const float*)d_in[15 + i];
    bb[i] = (const float*)d_in[24 + i];
  }
  const float* w_agg = (const float*)d_in[33];
  const float* g_agg = (const float*)d_in[34];
  const float* b_agg = (const float*)d_in[35];

  float* out_sampled = (float*)d_out;                    // (4,1024,3)
  float* out_y       = (float*)d_out + 12288;            // (4,128,1024)
  float* out_cls     = (float*)d_out + 12288 + 524288;   // (4,3,16384)

  char* ws = (char*)d_ws;
  unsigned* keys  = (unsigned*)ws;                // 256KB
  int* sidx       = (int*)(ws + 262144);          // 16KB
  int* gidx0      = (int*)(ws + 278528);          // 256KB (ns=16)
  int* gidx1      = (int*)(ws + 540672);          // 512KB (ns=32)
  int* gidx2      = (int*)(ws + 1064960);         // 1MB   (ns=64)
  _Float16* cat   = (_Float16*)(ws + 2113536);    // 3MB (4096 x 384 f16)

  conf_kernel<<<1024, 128, 0, stream>>>(feats, w_conf1, g_conf1, b_conf1,
                                        w_conf2, out_cls, keys);
  select_kernel<<<4, 1024, 0, stream>>>(points, keys, sidx, out_sampled);
  ball_kernel<<<512, 256, 0, stream>>>(points, out_sampled, gidx0, 16, 0.04f);
  ball_kernel<<<512, 256, 0, stream>>>(points, out_sampled, gidx1, 32, 0.16f);
  ball_kernel<<<512, 256, 0, stream>>>(points, out_sampled, gidx2, 64, 0.64f);

  size_t smBase = 25600;
  branch_kernel<16, 64><<<1024, 128, smBase + 4u * 2 * 16 * 96 * 2, stream>>>(
      points, feats, out_sampled, gidx0, bw[0], bw[1], bw[2], bg[0], bb[0],
      bg[1], bb[1], bg[2], bb[2], cat, 0);
  branch_kernel<32, 64><<<1024, 128, smBase + 4u * 2 * 32 * 96 * 2, stream>>>(
      points, feats, out_sampled, gidx1, bw[3], bw[4], bw[5], bg[3], bb[3],
      bg[4], bb[4], bg[5], bb[5], cat, 128);
  branch_kernel<64, 96><<<1024, 128, smBase + 4u * 2 * 64 * 96 * 2, stream>>>(
      points, feats, out_sampled, gidx2, bw[6], bw[7], bw[8], bg[6], bb[6],
      bg[7], bb[7], bg[8], bb[8], cat, 256);

  agg_kernel<<<64, 128, 98304 + 1024, stream>>>(cat, w_agg, g_agg, b_agg, out_y);
}
